// JODIE_82429012344975
// MI455X (gfx1250) — matrix operand order repaired and checked by reading, not compiled, not run on hardware
//
#include <hip/hip_runtime.h>

// ---------------------------------------------------------------------------
// JODIE temporal-interaction scan for MI455X (gfx1250), wave32 + WMMA bf16.
//
// Strategy: single persistent 512-thread (16-wave) workgroup runs the 2048
// sequential events. Matvecs are done with v_wmma_f32_16x16x32_bf16 where the
// input vector is replicated across all 16 B columns; fused RNN weights
// ([W_ih | pad | W_hh], K padded to 416) live in LDS (2 x 104KB), W_pred
// (256x640 bf16, 320KB) streams from L2 each step. Weights are pre-converted
// to fragment-ordered bf16 by prep kernels so every A/B fragment is two
// aligned 16-byte loads per lane.
// ---------------------------------------------------------------------------

typedef __bf16 bf16_t;
typedef bf16_t v16bf __attribute__((ext_vector_type(16)));
typedef float  v8f   __attribute__((ext_vector_type(8)));

#define D_       128
#define E_       2048
#define PRED_M   256
#define PRED_KP  640
#define PRED_KT  20      // 640/32
#define RNN_KP   416     // 257 (+31 pad) + 128, multiple of 32
#define RNN_KT   13      // 416/32
#define RIN      257
#define NTHREADS 512

// round-to-nearest-even f32 -> bf16 bits
__device__ __forceinline__ unsigned short f2bf(float x) {
  unsigned u = __builtin_bit_cast(unsigned, x);
  u += 0x7FFFu + ((u >> 16) & 1u);
  return (unsigned short)(u >> 16);
}

struct Frag32B { uint4 lo, hi; };
__device__ __forceinline__ v16bf frag_ld(const unsigned short* p) {
  Frag32B f;
  f.lo = *(const uint4*)(p);
  f.hi = *(const uint4*)(p + 8);
  return __builtin_bit_cast(v16bf, f);
}

__device__ __forceinline__ float wave_sum(float v) {
#pragma unroll
  for (int o = 16; o > 0; o >>= 1) v += __shfl_xor(v, o, 32);
  return v;
}

// --------------------------- prep kernels ----------------------------------

// d_out[0 .. N*D) = embedding copy; d_out[N*D] = 0 (loss slot)
__global__ void init_out_kernel(const float* __restrict__ emb,
                                float* __restrict__ out, long long n) {
  long long i = (long long)blockIdx.x * blockDim.x + threadIdx.x;
  if (i < n) out[i] = emb[i];
  else if (i == n) out[i] = 0.0f;
}

// W_pred (256x640 row-major f32) -> fragment-ordered bf16.
// Storage index: ((mt*PRED_KT + kt)*32 + lane)*16 + i
// Element map (A operand, contiguous k, same order as the B loads): h=lane/16, row=mt*16+(lane&15),
//   k = kt*32 + 16h + i
__global__ void prep_pred_kernel(const float* __restrict__ W,
                                 unsigned short* __restrict__ out) {
  int idx = blockIdx.x * blockDim.x + threadIdx.x;
  const int total = (PRED_M / 16) * PRED_KT * 32 * 16;
  if (idx >= total) return;
  int i    = idx & 15;
  int rest = idx >> 4;
  int lane = rest & 31;
  int tile = rest >> 5;
  int kt   = tile % PRED_KT;
  int mt   = tile / PRED_KT;
  int h    = lane >> 4;
  int row  = mt * 16 + (lane & 15);
  int k    = kt * 32 + 16 * h + i;
  out[idx] = f2bf(W[row * PRED_KP + k]);
}

// Fused RNN weight [W_ih (128x257) | zeros (31) | W_hh (128x128)] -> frag bf16
__global__ void prep_rnn_kernel(const float* __restrict__ Wih,
                                const float* __restrict__ Whh,
                                unsigned short* __restrict__ out) {
  int idx = blockIdx.x * blockDim.x + threadIdx.x;
  const int total = (D_ / 16) * RNN_KT * 32 * 16;
  if (idx >= total) return;
  int i    = idx & 15;
  int rest = idx >> 4;
  int lane = rest & 31;
  int tile = rest >> 5;
  int kt   = tile % RNN_KT;
  int mt   = tile / RNN_KT;
  int h    = lane >> 4;
  int row  = mt * 16 + (lane & 15);
  int k    = kt * 32 + 16 * h + i;
  float v;
  if (k < RIN)        v = Wih[row * RIN + k];
  else if (k < 288)   v = 0.0f;
  else                v = Whh[row * D_ + (k - 288)];
  out[idx] = f2bf(v);
}

// --------------------------- main kernel -----------------------------------

// LDS layout (bytes):
#define L_WU     0                         // 8*13*512 ushorts = 106496 B
#define L_WL     106496                    // 106496 B
#define L_META   212992                    // 640 ushorts = 1280 B (32B aligned)
#define L_XU     214272                    // 416 ushorts = 832 B
#define L_XL     215104                    // 416 ushorts = 832 B
#define L_F      215936                    // f32 region (16B aligned)
#define L_U      (L_F + 0)
#define L_L      (L_F + 512)
#define L_P      (L_F + 1024)
#define L_KG     (L_F + 1536)
#define L_SP     (L_F + 2048)
#define L_SU     (L_F + 2560)
#define L_SL     (L_F + 3072)
#define L_PRED   (L_F + 3584)              // 256 f32
#define L_PREU   (L_F + 4608)              // 128 f32
#define L_PREL   (L_F + 5120)              // 128 f32
#define L_RED    (L_F + 5632)              // 64 f32 (4 slots x 16 waves)
#define L_SCAL   (L_F + 5888)              // 8 f32
#define L_LOSS   (L_F + 5920)              // 1 f32
#define SMEM_BYTES (L_F + 5952)

extern "C" __global__ void __launch_bounds__(NTHREADS, 1)
jodie_main_kernel(const float* __restrict__ statics,
                  const float* __restrict__ kg,
                  const float* __restrict__ b_ih_u, const float* __restrict__ b_hh_u,
                  const float* __restrict__ b_ih_l, const float* __restrict__ b_hh_l,
                  const float* __restrict__ w_proj, const float* __restrict__ b_proj,
                  const float* __restrict__ ln_g,   const float* __restrict__ ln_b,
                  const float* __restrict__ b_pred,
                  const float* __restrict__ delta_u, const float* __restrict__ delta_l,
                  const int* __restrict__ idx_user, const int* __restrict__ idx_loca,
                  const int* __restrict__ idx_prev, const int* __restrict__ idx_know,
                  const int* __restrict__ num_users_p,
                  const unsigned short* __restrict__ wpred_f,
                  const unsigned short* __restrict__ wu_f,
                  const unsigned short* __restrict__ wl_f,
                  float* __restrict__ out, long long loss_off) {
  extern __shared__ char smem[];
  unsigned short* sWu   = (unsigned short*)(smem + L_WU);
  unsigned short* sWl   = (unsigned short*)(smem + L_WL);
  unsigned short* sMeta = (unsigned short*)(smem + L_META);
  unsigned short* sXu   = (unsigned short*)(smem + L_XU);
  unsigned short* sXl   = (unsigned short*)(smem + L_XL);
  float* sU    = (float*)(smem + L_U);
  float* sL    = (float*)(smem + L_L);
  float* sP    = (float*)(smem + L_P);
  float* sKG   = (float*)(smem + L_KG);
  float* sSP   = (float*)(smem + L_SP);
  float* sSU   = (float*)(smem + L_SU);
  float* sSL   = (float*)(smem + L_SL);
  float* sPred = (float*)(smem + L_PRED);
  float* sPreU = (float*)(smem + L_PREU);
  float* sPreL = (float*)(smem + L_PREL);
  float* sRed  = (float*)(smem + L_RED);
  float* sScal = (float*)(smem + L_SCAL);
  float* sLoss = (float*)(smem + L_LOSS);

  const int tid  = threadIdx.x;
  const int lane = tid & 31;
  const int wv   = tid >> 5;          // 0..15
  const int seg  = tid >> 7;          // 0..3
  const int c    = tid & 127;
  const int bofs = lane & 16;         // 0 or 16: B-fragment K half per lane
  const int NU   = *num_users_p;

  // ---- load fused RNN weights into LDS (once) ----
  {
    const int U4 = (8 * RNN_KT * 32 * 16) / 8;   // 6656 uint4 per matrix
    uint4* du4 = (uint4*)sWu;
    uint4* dl4 = (uint4*)sWl;
    const uint4* su4 = (const uint4*)wu_f;
    const uint4* sl4 = (const uint4*)wl_f;
    for (int i = tid; i < U4; i += NTHREADS) {
      du4[i] = su4[i];
      dl4[i] = sl4[i];
    }
  }
  if (tid < 31) { sXu[RIN + tid] = 0; sXl[RIN + tid] = 0; }  // pad cols 257..287
  if (tid == 0) *sLoss = 0.0f;
  __syncthreads();

  for (int t = 0; t < E_; ++t) {
    const int   iu = idx_user[t];
    const int   il = idx_loca[t] + NU;
    const int   ip = idx_prev[t] + NU;
    const int   ik = idx_know[t];
    const float du = delta_u[t];
    const float dl = delta_l[t];

    // ---- gather rows (embedding read from evolving copy in `out`) ----
    if (seg == 0)      sU[c]  = out[(long long)iu * D_ + c];
    else if (seg == 1) sL[c]  = out[(long long)il * D_ + c];
    else if (seg == 2) sP[c]  = out[(long long)ip * D_ + c];
    else               sKG[c] = kg[(long long)ik * D_ + c];
    if (seg == 0)      sSP[c] = statics[(long long)ip * D_ + c];
    else if (seg == 1) sSU[c] = statics[(long long)iu * D_ + c];
    else if (seg == 2) sSL[c] = statics[(long long)il * D_ + c];
    __syncthreads();

    // ---- LayerNorm stats over concat(p, kg) [256] ----
    float xv = 0.0f;
    if (tid < 256) xv = (tid < 128) ? sP[tid] : sKG[tid - 128];
    float s1 = wave_sum(xv);
    float s2 = wave_sum(xv * xv);
    if (lane == 0) { sRed[wv] = s1; sRed[16 + wv] = s2; }
    __syncthreads();
    if (tid == 0) {
      float a = 0.0f, b = 0.0f;
      for (int i = 0; i < 16; ++i) { a += sRed[i]; b += sRed[16 + i]; }
      float mu = a * (1.0f / 256.0f);
      float var = b * (1.0f / 256.0f) - mu * mu;
      sScal[0] = mu;
      sScal[1] = rsqrtf(var + 1e-5f);
    }
    __syncthreads();
    const float mu = sScal[0], rstd = sScal[1];

    // ---- build bf16 input vectors: meta[640], x_u[416], x_l[416] ----
    if (tid < 128) {
      float proj = sU[tid] * (1.0f + w_proj[tid] * du + b_proj[tid]);
      sMeta[tid] = f2bf(proj);
      sMeta[384 + tid] = f2bf(sSP[tid]);
      sMeta[512 + tid] = f2bf(sSU[tid]);
      // x_u = [l, kg, du, pad, u] ; x_l = [u, kg, dl, pad, l]
      sXu[tid]       = f2bf(sL[tid]);
      sXu[128 + tid] = f2bf(sKG[tid]);
      sXu[288 + tid] = f2bf(sU[tid]);
      sXl[tid]       = f2bf(sU[tid]);
      sXl[128 + tid] = f2bf(sKG[tid]);
      sXl[288 + tid] = f2bf(sL[tid]);
    }
    if (tid < 256) {
      float lnv = (xv - mu) * rstd * ln_g[tid] + ln_b[tid];
      sMeta[128 + tid] = f2bf(lnv);
    }
    if (tid == 0) { sXu[256] = f2bf(du); sXl[256] = f2bf(dl); }
    __syncthreads();

    // ---- WMMA phase 1: pred = W_pred @ meta (16 M-tiles, one per wave) ----
    {
      v8f acc = {};
      const unsigned short* abase =
          wpred_f + ((size_t)(wv * PRED_KT) * 32 + lane) * 16;
#pragma unroll 4
      for (int kt = 0; kt < PRED_KT; ++kt) {
        v16bf a = frag_ld(abase + kt * 512);
        v16bf b = frag_ld(sMeta + kt * 32 + bofs);  // vector replicated in N
        acc = __builtin_amdgcn_wmma_f32_16x16x32_bf16(
            false, a, false, b, (short)0, acc, false, false);
      }
      if (lane == 0) {
#pragma unroll
        for (int r = 0; r < 8; ++r) sPred[wv * 16 + r] = acc[r];
      } else if (lane == 16) {
#pragma unroll
        for (int r = 0; r < 8; ++r) sPred[wv * 16 + 8 + r] = acc[r];
      }
    }
    // ---- WMMA phase 2: fused RNN matvecs (waves 0-7: user, 8-15: loca) ----
    {
      v8f acc = {};
      const int mt = wv & 7;
      const unsigned short* W  = (wv < 8) ? sWu : sWl;
      const unsigned short* xb = (wv < 8) ? sXu : sXl;
      const unsigned short* abase = W + ((mt * RNN_KT) * 32 + lane) * 16;
#pragma unroll
      for (int kt = 0; kt < RNN_KT; ++kt) {
        v16bf a = frag_ld(abase + kt * 512);
        v16bf b = frag_ld(xb + kt * 32 + bofs);
        acc = __builtin_amdgcn_wmma_f32_16x16x32_bf16(
            false, a, false, b, (short)0, acc, false, false);
      }
      float* dst = (wv < 8) ? sPreU : sPreL;
      if (lane == 0) {
#pragma unroll
        for (int r = 0; r < 8; ++r) dst[mt * 16 + r] = acc[r];
      } else if (lane == 16) {
#pragma unroll
        for (int r = 0; r < 8; ++r) dst[mt * 16 + 8 + r] = acc[r];
      }
    }
    __syncthreads();

    // ---- losses + activations ----
    float pd = 0.0f, hu = 0.0f, hl = 0.0f;
    if (tid < 256) {
      float p  = sPred[tid] + b_pred[tid];
      float tg = (tid < 128) ? sL[tid] : sSL[tid - 128];
      float d  = p - tg;
      pd = d * d;
    }
    if (tid < 128) {
      hu = tanhf(sPreU[tid] + b_ih_u[tid] + b_hh_u[tid]);
      hl = tanhf(sPreL[tid] + b_ih_l[tid] + b_hh_l[tid]);
    }
    float r0 = wave_sum(pd), r1 = wave_sum(hu * hu), r2 = wave_sum(hl * hl);
    if (lane == 0) { sRed[wv] = r0; sRed[16 + wv] = r1; sRed[32 + wv] = r2; }
    __syncthreads();
    if (tid == 0) {
      float a = 0.0f, b = 0.0f, cc = 0.0f;
      for (int i = 0; i < 16; ++i) {
        a += sRed[i]; b += sRed[16 + i]; cc += sRed[32 + i];
      }
      sScal[0] = a * (1.0f / 256.0f);              // loss_pred
      sScal[1] = 1.0f / fmaxf(sqrtf(b), 1e-12f);   // 1/||new_u||
      sScal[2] = 1.0f / fmaxf(sqrtf(cc), 1e-12f);  // 1/||new_l||
    }
    __syncthreads();
    const float inu = sScal[1], inl = sScal[2];
    float du2 = 0.0f, dl2 = 0.0f;
    if (tid < 128) {
      float nu = hu * inu, nl = hl * inl;
      out[(long long)iu * D_ + tid] = nu;
      out[(long long)il * D_ + tid] = nl;
      float a = nu - sU[tid]; du2 = a * a;
      float b = nl - sL[tid]; dl2 = b * b;
    }
    r0 = wave_sum(du2); r1 = wave_sum(dl2);
    if (lane == 0) { sRed[wv] = r0; sRed[16 + wv] = r1; }
    __syncthreads();
    if (tid == 0) {
      float a = 0.0f, b = 0.0f;
      for (int i = 0; i < 16; ++i) { a += sRed[i]; b += sRed[16 + i]; }
      *sLoss += sScal[0] + a * (1.0f / 128.0f) + b * (1.0f / 128.0f);
    }
    __threadfence();   // make emb updates visible to next iteration's gathers
    __syncthreads();
  }

  if (tid == 0) out[loss_off] = *sLoss;
}

// --------------------------- launch ----------------------------------------

extern "C" void kernel_launch(void* const* d_in, const int* in_sizes, int n_in,
                              void* d_out, int out_size, void* d_ws, size_t ws_size,
                              hipStream_t stream) {
  (void)in_sizes; (void)n_in; (void)out_size; (void)ws_size;
  const float* emb     = (const float*)d_in[0];
  const float* statics = (const float*)d_in[1];
  const float* kgp     = (const float*)d_in[2];
  const float* W_ih_u  = (const float*)d_in[3];
  const float* b_ih_u  = (const float*)d_in[4];
  const float* W_hh_u  = (const float*)d_in[5];
  const float* b_hh_u  = (const float*)d_in[6];
  const float* W_ih_l  = (const float*)d_in[7];
  const float* b_ih_l  = (const float*)d_in[8];
  const float* W_hh_l  = (const float*)d_in[9];
  const float* b_hh_l  = (const float*)d_in[10];
  const float* w_proj  = (const float*)d_in[11];
  const float* b_proj  = (const float*)d_in[12];
  const float* ln_g    = (const float*)d_in[13];
  const float* ln_b    = (const float*)d_in[14];
  const float* W_pred  = (const float*)d_in[15];
  const float* b_pred  = (const float*)d_in[16];
  const float* delta_u = (const float*)d_in[17];
  const float* delta_l = (const float*)d_in[18];
  const int* idx_user  = (const int*)d_in[19];
  const int* idx_loca  = (const int*)d_in[20];
  const int* idx_prev  = (const int*)d_in[21];
  const int* idx_know  = (const int*)d_in[22];
  const int* num_users = (const int*)d_in[23];
  float* out = (float*)d_out;

  // workspace carve: fragment-ordered bf16 weights
  unsigned short* wpred_f = (unsigned short*)d_ws;                    // 327680 B
  unsigned short* wu_f    = (unsigned short*)((char*)d_ws + 327680);  // 106496 B
  unsigned short* wl_f    = (unsigned short*)((char*)d_ws + 434176);  // 106496 B

  const long long NE = 150000LL * 128LL;  // N*D

  {
    long long total = NE + 1;
    int blocks = (int)((total + 255) / 256);
    init_out_kernel<<<blocks, 256, 0, stream>>>(emb, out, NE);
  }
  {
    const int total = (PRED_M / 16) * PRED_KT * 32 * 16;  // 163840
    prep_pred_kernel<<<(total + 255) / 256, 256, 0, stream>>>(W_pred, wpred_f);
  }
  {
    const int total = (D_ / 16) * RNN_KT * 32 * 16;       // 53248
    prep_rnn_kernel<<<(total + 255) / 256, 256, 0, stream>>>(W_ih_u, W_hh_u, wu_f);
    prep_rnn_kernel<<<(total + 255) / 256, 256, 0, stream>>>(W_ih_l, W_hh_l, wl_f);
  }

  jodie_main_kernel<<<1, NTHREADS, SMEM_BYTES, stream>>>(
      statics, kgp, b_ih_u, b_hh_u, b_ih_l, b_hh_l, w_proj, b_proj,
      ln_g, ln_b, b_pred, delta_u, delta_l,
      idx_user, idx_loca, idx_prev, idx_know, num_users,
      wpred_f, wu_f, wl_f, out, NE);
}
